// GAHeM_51823075393820
// MI455X (gfx1250) — matrix operand order repaired and checked by reading, not compiled, not run on hardware
//
#include <hip/hip_runtime.h>

// ---------------------------------------------------------------------------
// MI455X (gfx1250, wave32) implementation.
//
// Heavy GEMMs -> v_wmma_f32_16x16x32_bf16 (fp32 accumulate). Tiles are staged
// into LDS in *fragment order* so each lane's WMMA operand is one contiguous
// 32-byte slice -> 2x ds_load_b128 per fragment (round-1 codegen showed 16x
// ds_load_u16 + 16x v_mov_b16 per fragment with naive tile layout).
// Edge softmax -> ordered-uint atomicMax + f32 atomicAdd.
// GRU restructured: one forward scan + 16 batched backward steps
// ((512,300)@(300,900) per step) -> critical path is 16 WMMA GEMMs.
// ---------------------------------------------------------------------------

typedef __attribute__((ext_vector_type(16))) __bf16 v16bf;
typedef __attribute__((ext_vector_type(8)))  float  v8f;

#define TILE_M 64
#define TILE_N 64
#define TILE_K 32

enum { EPI_NONE = 0, EPI_BIAS = 1, EPI_BIAS_RELU = 2, EPI_BIAS_SIGMOID = 3 };

__device__ __forceinline__ unsigned int pack_bf16(float a, float b) {
    union { __bf16 h[2]; unsigned int u; } p;
    p.h[0] = (__bf16)a; p.h[1] = (__bf16)b;
    return p.u;
}

// ------------------------------ WMMA GEMM ----------------------------------
// C(MxN) = A(MxK) @ B  (+bias, +act).  transB==0: B is (K,N) row-major.
// transB==1: B stored (N,K) row-major (i.e. compute A @ B^T of stored W).
__global__ __launch_bounds__(128)
void wmma_gemm_kernel(const float* __restrict__ A, const float* __restrict__ B,
                      const float* __restrict__ bias, float* __restrict__ C,
                      int M, int N, int K, int transB, int epilogue)
{
    // Fragment-ordered staging: [wave/ntile][lane][bf16-pair], 4 KB each.
    // A lane's operand = AfragU[w][lane][0..7] = 32 contiguous bytes.
    __shared__ __align__(16) unsigned int AfragU[4][32][8];
    __shared__ __align__(16) unsigned int BfragU[4][32][8];

    const int tid  = threadIdx.x;
    const int wave = tid >> 5;          // 4 waves / block (wave32)
    const int lane = tid & 31;
    const int half = lane >> 4;
    const int l16g = lane & 15;
    const int row0 = blockIdx.y * TILE_M;
    const int col0 = blockIdx.x * TILE_N;

    v8f acc[4] = {};

    const int kTiles = (K + TILE_K - 1) / TILE_K;
    for (int kt = 0; kt < kTiles; ++kt) {
        const int k0 = kt * TILE_K;

        // ---- stage A: 64 rows x 16 k-pairs; contiguous fp32 pair -> packed
        //      bf16 dword stored at its ISA 7.12.2 fragment slot.
        for (int i = tid; i < 64 * 16; i += 128) {
            int r  = i >> 4;            // local row 0..63
            int c0 = (i & 15) << 1;     // even local k
            int gr = row0 + r, gk = k0 + c0;
            float v0 = 0.f, v1 = 0.f;
            if (gr < M) {
                if (gk + 1 < K) {
                    v0 = A[(size_t)gr * K + gk];
                    v1 = A[(size_t)gr * K + gk + 1];
                    if (gk + TILE_K < K)
                        __builtin_prefetch(&A[(size_t)gr * K + gk + TILE_K], 0, 1);
                } else if (gk < K) {
                    v0 = A[(size_t)gr * K + gk];
                }
            }
            // A 16x32 bf16 layout: lane = ((c>>3)&1)*16 + (r&15),
            // element e = (c>>4)*8 + (c&7)   (pairs stay adjacent).
            int w   = r >> 4, l16 = r & 15;
            int ln  = (((c0 >> 3) & 1) << 4) + l16;
            int pr  = ((c0 >> 4) * 8 + (c0 & 7)) >> 1;
            AfragU[w][ln][pr] = pack_bf16(v0, v1);
        }
        // ---- stage B: 16 k-pairs x 64 n.
        for (int i = tid; i < 16 * 64; i += 128) {
            int k0l = (i >> 6) << 1;    // even local k
            int n   = i & 63;
            int gk  = k0 + k0l, gc = col0 + n;
            float v0 = 0.f, v1 = 0.f;
            if (gc < N) {
                if (transB) {           // contiguous along k
                    if (gk + 1 < K) { v0 = B[(size_t)gc*K + gk]; v1 = B[(size_t)gc*K + gk + 1]; }
                    else if (gk < K) v0 = B[(size_t)gc*K + gk];
                } else {
                    if (gk < K)       v0 = B[(size_t)gk*N + gc];
                    if (gk + 1 < K)   v1 = B[(size_t)(gk+1)*N + gc];
                }
            }
            // B 32x16 bf16 layout: lane = ((k>>3)&1)*16 + (n&15), element e = (k>>4)*8 + (k&7).
            int j  = n >> 4, l16 = n & 15;
            int ln = (((k0l >> 3) & 1) << 4) + l16;
            int pr = ((k0l >> 4) * 8 + (k0l & 7)) >> 1;
            BfragU[j][ln][pr] = pack_bf16(v0, v1);
        }
        __syncthreads();

        // Operand fetch: one contiguous 32B slice per fragment (2x ds_load_b128).
        v16bf afrag = *(const v16bf*)(&AfragU[wave][lane][0]);
        #pragma unroll
        for (int j = 0; j < 4; ++j) {
            v16bf bfrag = *(const v16bf*)(&BfragU[j][lane][0]);
            acc[j] = __builtin_amdgcn_wmma_f32_16x16x32_bf16(
                false, afrag, false, bfrag, (short)0, acc[j], false, false);
        }
        __syncthreads();
    }

    // C layout: VGPR e -> M = e + 8*half ; N = lane&15 (ISA 7.12.2)
    #pragma unroll
    for (int j = 0; j < 4; ++j) {
        #pragma unroll
        for (int e = 0; e < 8; ++e) {
            int gr = row0 + 16 * wave + (half << 3) + e;
            int gc = col0 + 16 * j + l16g;
            if (gr < M && gc < N) {
                float v = acc[j][e];
                if (epilogue != EPI_NONE)       v += bias[gc];
                if (epilogue == EPI_BIAS_RELU)  v = fmaxf(v, 0.0f);
                else if (epilogue == EPI_BIAS_SIGMOID) v = 1.0f / (1.0f + expf(-v));
                C[(size_t)gr * N + gc] = v;
            }
        }
    }
}

// --------------------------- small helpers ---------------------------------
__global__ void fill_f32_kernel(float* p, float v, size_t n) {
    size_t i = (size_t)blockIdx.x * blockDim.x + threadIdx.x;
    if (i < n) p[i] = v;
}
__global__ void fill_u32_kernel(unsigned int* p, unsigned int v, size_t n) {
    size_t i = (size_t)blockIdx.x * blockDim.x + threadIdx.x;
    if (i < n) p[i] = v;
}

__device__ __forceinline__ unsigned int f2ord(float f) {
    unsigned int u = __float_as_uint(f);
    return (u & 0x80000000u) ? ~u : (u | 0x80000000u);
}
__device__ __forceinline__ float ord2f(unsigned int u) {
    return (u & 0x80000000u) ? __uint_as_float(u ^ 0x80000000u)
                             : __uint_as_float(~u);
}
__device__ __forceinline__ float sigm(float x) { return 1.0f / (1.0f + expf(-x)); }

// ------------------------------- GAT ---------------------------------------
__global__ void gat_attn_coef_kernel(const float* __restrict__ xw,
                                     const float* __restrict__ a_s,
                                     const float* __restrict__ a_d,
                                     float* al, float* ar, int n) {
    int i = blockIdx.x * blockDim.x + threadIdx.x;     // n*4
    if (i >= n * 4) return;
    int node = i >> 2, h = i & 3;
    const float* x = xw + (size_t)node * 256 + h * 64;
    float sl = 0.f, sr = 0.f;
    for (int c = 0; c < 64; ++c) { float v = x[c]; sl += v * a_s[h*64+c]; sr += v * a_d[h*64+c]; }
    al[i] = sl; ar[i] = sr;
}

__device__ __forceinline__ void edge_sd(const int* src, const int* dst, int e, int E,
                                        int& s, int& d) {
    if (e < E) { s = src[e]; d = dst[e]; } else { s = e - E; d = e - E; }   // self-loops
}

__global__ void gat_edge_max_kernel(const int* src, const int* dst, int E, int n,
                                    const float* al, const float* ar, unsigned int* nmax) {
    int i = blockIdx.x * blockDim.x + threadIdx.x;     // (E+n)*4
    if (i >= (E + n) * 4) return;
    int e = i >> 2, h = i & 3, s, d;
    edge_sd(src, dst, e, E, s, d);
    float x = al[s*4+h] + ar[d*4+h];
    float lg = (x >= 0.f) ? x : 0.2f * x;              // leaky_relu(0.2)
    atomicMax(&nmax[d*4+h], f2ord(lg));
}

__global__ void gat_edge_sum_kernel(const int* src, const int* dst, int E, int n,
                                    const float* al, const float* ar,
                                    const unsigned int* nmax, float* nsum) {
    int i = blockIdx.x * blockDim.x + threadIdx.x;
    if (i >= (E + n) * 4) return;
    int e = i >> 2, h = i & 3, s, d;
    edge_sd(src, dst, e, E, s, d);
    float x = al[s*4+h] + ar[d*4+h];
    float lg = (x >= 0.f) ? x : 0.2f * x;
    atomicAdd(&nsum[d*4+h], expf(lg - ord2f(nmax[d*4+h])));
}

__global__ void gat_edge_agg_kernel(const int* src, const int* dst, int E, int n,
                                    const float* al, const float* ar,
                                    const unsigned int* nmax, const float* nsum,
                                    const float* __restrict__ xw, float* agg) {
    int i = blockIdx.x * blockDim.x + threadIdx.x;
    if (i >= (E + n) * 4) return;
    int e = i >> 2, h = i & 3, s, d;
    edge_sd(src, dst, e, E, s, d);
    float x = al[s*4+h] + ar[d*4+h];
    float lg = (x >= 0.f) ? x : 0.2f * x;
    float ev = expf(lg - ord2f(nmax[d*4+h]));
    float alpha = ev / (nsum[d*4+h] + 1e-16f);
    const float* xr = xw + (size_t)s * 256 + h * 64;
    float*      orow = agg + (size_t)d * 256 + h * 64;
    for (int c = 0; c < 64; ++c) atomicAdd(&orow[c], alpha * xr[c]);
}

__global__ void gat_finalize_kernel(float* p, const float* bias, size_t n) {
    size_t i = (size_t)blockIdx.x * blockDim.x + threadIdx.x;
    if (i < n) p[i] = fmaxf(p[i] + bias[i & 255], 0.0f);   // +b, relu
}

// --------------------------- semantic attention ----------------------------
__global__ void semantic_kernel(const float* __restrict__ embs, int g0, int J,
                                const float* w, const float* bptr,
                                float* out, int colOff, int n) {
    int node = blockIdx.x * blockDim.x + threadIdx.x;
    if (node >= n) return;
    float beta[3]; float mx = -1e30f; float b = bptr[0];
    for (int j = 0; j < J; ++j) {
        const float* m = embs + ((size_t)(g0 + j) * n + node) * 256;
        float s = b;
        for (int c = 0; c < 256; ++c) s += m[c] * w[c];
        s = fmaxf(s, 0.f);
        beta[j] = s; mx = fmaxf(mx, s);
    }
    float den = 0.f;
    for (int j = 0; j < J; ++j) { beta[j] = expf(beta[j] - mx); den += beta[j]; }
    for (int j = 0; j < J; ++j) beta[j] /= den;
    for (int c = 0; c < 256; ++c) {
        float s = 0.f;
        for (int j = 0; j < J; ++j)
            s += beta[j] * embs[((size_t)(g0 + j) * n + node) * 256 + c];
        out[(size_t)node * 512 + colOff + c] = s / (float)J;   // jnp.mean over axis=1
    }
}

// ---------------------- layout shuffles / GRU / output ---------------------
__global__ void transpose_bt_kernel(const float* xt, float* day) {   // (B,T,512)->(T,B,512)
    int i = blockIdx.x * blockDim.x + threadIdx.x;
    if (i >= 512 * 512) return;
    int r = i >> 9, c = i & 511, t = r >> 5, b = r & 31;
    day[i] = xt[(((size_t)b * 16) + t) * 512 + c];
}

__global__ void gru_fwd_step_kernel(const float* __restrict__ gi,  // (16*32,900) row t*32+b
                                    const float* __restrict__ gh,  // (32,900)
                                    float* h, float* fh, int t) {
    int i = blockIdx.x * blockDim.x + threadIdx.x;
    if (i >= 32 * 300) return;
    int b = i / 300, c = i % 300;
    const float* gir = gi + ((size_t)t * 32 + b) * 900;
    const float* ghr = gh + (size_t)b * 900;
    float r  = sigm(gir[c]       + ghr[c]);
    float z  = sigm(gir[300 + c] + ghr[300 + c]);
    float ng = tanhf(gir[600 + c] + r * ghr[600 + c]);
    float hn = (1.f - z) * ng + z * h[i];
    h[i] = hn;
    fh[((size_t)t * 32 + b) * 300 + c] = hn;
}

// 16 backward chains batched: chain i at step s processes original t = i-s.
__global__ void gru_bwd_step_kernel(const float* __restrict__ gi,  // (16*32,900) row t*32+b
                                    const float* __restrict__ gh,  // (512,900) row i*32+b
                                    float* Hb, float* rh, int s) {
    int idx = blockIdx.x * blockDim.x + threadIdx.x;
    if (idx >= 16 * 32 * 300) return;
    int i = idx / (32 * 300);
    if (i < s) return;
    int rem = idx % (32 * 300), b = rem / 300, c = rem % 300, t = i - s;
    const float* gir = gi + ((size_t)t * 32 + b) * 900;
    const float* ghr = gh + ((size_t)i * 32 + b) * 900;
    float r  = sigm(gir[c]       + ghr[c]);
    float z  = sigm(gir[300 + c] + ghr[300 + c]);
    float ng = tanhf(gir[600 + c] + r * ghr[600 + c]);
    size_t hidx = ((size_t)i * 32 + b) * 300 + c;
    float hn = (1.f - z) * ng + z * Hb[hidx];
    Hb[hidx] = hn;
    rh[(((size_t)i * 16 + t) * 32 + b) * 300 + c] = hn;
}

// Temporal attention per (prefix i, batch b): softmax over t, mean pool, concat last.
__global__ __launch_bounds__(128)
void time_attn_kernel(const float* __restrict__ fh, const float* __restrict__ rh,
                      const float* __restrict__ attn_w, const float* __restrict__ attn_b,
                      float* __restrict__ h_t) {
    int blk = blockIdx.x, i = blk >> 5, b = blk & 31, tid = threadIdx.x;
    __shared__ float red[128];
    __shared__ float alpha[16];
    for (int t = 0; t <= i; ++t) {
        float p = 0.f;
        for (int c = tid; c < 600; c += 128) {
            float v = (c < 300) ? fh[((size_t)t * 32 + b) * 300 + c]
                                : rh[(((size_t)i * 16 + t) * 32 + b) * 300 + (c - 300)];
            p += v * attn_w[c];
        }
        red[tid] = p; __syncthreads();
        for (int st = 64; st > 0; st >>= 1) { if (tid < st) red[tid] += red[tid + st]; __syncthreads(); }
        if (tid == 0) alpha[t] = red[0] + attn_b[0];
        __syncthreads();
    }
    if (tid == 0) {
        float mx = -1e30f;
        for (int t = 0; t <= i; ++t) mx = fmaxf(mx, alpha[t]);
        float den = 0.f;
        for (int t = 0; t <= i; ++t) { alpha[t] = expf(alpha[t] - mx); den += alpha[t]; }
        for (int t = 0; t <= i; ++t) alpha[t] /= den;
    }
    __syncthreads();
    float inv = 1.0f / (float)(i + 1);
    for (int c = tid; c < 600; c += 128) {
        float s = 0.f;
        for (int t = 0; t <= i; ++t) {
            float v = (c < 300) ? fh[((size_t)t * 32 + b) * 300 + c]
                                : rh[(((size_t)i * 16 + t) * 32 + b) * 300 + (c - 300)];
            s += alpha[t] * v;
        }
        float last = (c < 300) ? fh[((size_t)i * 32 + b) * 300 + c]
                               : rh[(((size_t)i * 16 + i) * 32 + b) * 300 + (c - 300)];
        h_t[(size_t)blk * 1200 + c] = s * inv;          // c_t (mean pooling)
        h_t[(size_t)blk * 1200 + 600 + c] = last;       // rnn_h[-1]
    }
}

__global__ void out_permute_kernel(const float* ytmp, float* y) {   // (T,B,500)->(B,T,500)
    int i = blockIdx.x * blockDim.x + threadIdx.x;
    if (i >= 512 * 500) return;
    int r = i / 500, o = i % 500, t = r >> 5, b = r & 31;
    y[((size_t)b * 16 + t) * 500 + o] = ytmp[i];
}

// ------------------------------- driver ------------------------------------
static inline void gemm(hipStream_t st, const float* A, const float* B, const float* bias,
                        float* C, int M, int N, int K, int transB, int epi) {
    dim3 g((N + TILE_N - 1) / TILE_N, (M + TILE_M - 1) / TILE_M), b(128);
    wmma_gemm_kernel<<<g, b, 0, st>>>(A, B, bias, C, M, N, K, transB, epi);
}
static inline void fillf(hipStream_t st, float* p, float v, size_t n) {
    fill_f32_kernel<<<(unsigned)((n + 255) / 256), 256, 0, st>>>(p, v, n);
}
static inline void fillu(hipStream_t st, unsigned int* p, unsigned int v, size_t n) {
    fill_u32_kernel<<<(unsigned)((n + 255) / 256), 256, 0, st>>>(p, v, n);
}

extern "C" void kernel_launch(void* const* d_in, const int* in_sizes, int n_in,
                              void* d_out, int out_size, void* d_ws, size_t ws_size,
                              hipStream_t stream)
{
    (void)n_in; (void)out_size; (void)ws_size;
    const float* x      = (const float*)d_in[0];
    const int*   edges  = (const int*)  d_in[1];
    const float* emb0   = (const float*)d_in[2];
    const float* W1     = (const float*)d_in[3];
    const float* a1s    = (const float*)d_in[4];
    const float* a1d    = (const float*)d_in[5];
    const float* b1     = (const float*)d_in[6];
    const float* W2     = (const float*)d_in[7];
    const float* a2s    = (const float*)d_in[8];
    const float* a2d    = (const float*)d_in[9];
    const float* b2     = (const float*)d_in[10];
    const float* sem_w  = (const float*)d_in[11];
    const float* sem_b  = (const float*)d_in[12];
    const float* Wih    = (const float*)d_in[13];
    const float* Whh    = (const float*)d_in[14];
    const float* bih    = (const float*)d_in[15];
    const float* bhh    = (const float*)d_in[16];
    const float* attn_w = (const float*)d_in[17];
    const float* attn_b = (const float*)d_in[18];
    const float* aout_w = (const float*)d_in[19];
    const float* aout_b = (const float*)d_in[20];
    const float* out_w  = (const float*)d_in[21];
    const float* out_b  = (const float*)d_in[22];

    const int NN = 20000;
    const int E  = in_sizes[1] / 10;            // 5*2*E
    const int Et4 = (E + NN) * 4;

    float* y        = (float*)d_out;            // (32,16,500)
    float* attn_out = y + 512 * 500;            // (20000,512) — second tuple output

    // -------- workspace carve-up (floats) --------
    float* ws = (float*)d_ws;  size_t o = 0;
    float* embs  = ws + o; o += 5UL * NN * 256;
    float* xw    = ws + o; o += (size_t)NN * 256;
    float* h2    = ws + o; o += (size_t)NN * 256;
    float* al    = ws + o; o += (size_t)NN * 4;
    float* ar    = ws + o; o += (size_t)NN * 4;
    unsigned int* nmax = (unsigned int*)(ws + o); o += (size_t)NN * 4;
    float* nsum  = ws + o; o += (size_t)NN * 4;
    float* xt    = ws + o; o += 512UL * 512;
    float* day   = ws + o; o += 512UL * 512;
    float* gif   = ws + o; o += 512UL * 900;
    float* gib   = ws + o; o += 512UL * 900;
    float* hstate= ws + o; o += 32UL * 300;
    float* ghf   = ws + o; o += 32UL * 900;
    float* fh    = ws + o; o += 16UL * 32 * 300;
    float* Hb    = ws + o; o += 512UL * 300;
    float* ghb   = ws + o; o += 512UL * 900;
    float* rh    = ws + o; o += 16UL * 16 * 32 * 300;
    float* ht    = ws + o; o += 512UL * 1200;
    float* houts = ws + o; o += 512UL * 512;
    float* ytmp  = ws + o; o += 512UL * 500;

    const unsigned ORD_NEG_INF = 0x007FFFFFu;   // f2ord(-inf)

    // ---------------- 5 GAT stacks ----------------
    for (int g = 0; g < 5; ++g) {
        const int* src = edges + (size_t)g * 2 * E;
        const int* dst = src + E;
        for (int layer = 0; layer < 2; ++layer) {
            const float* hin  = layer ? h2 : emb0;
            const int    Kin  = layer ? 256 : 64;
            const float* W    = layer ? (W2 + (size_t)g * 256 * 256) : (W1 + (size_t)g * 64 * 256);
            const float* as_  = (layer ? a2s : a1s) + (size_t)g * 256;
            const float* ad_  = (layer ? a2d : a1d) + (size_t)g * 256;
            const float* bias = (layer ? b2  : b1 ) + (size_t)g * 256;
            float* outbuf     = layer ? (embs + (size_t)g * NN * 256) : h2;

            gemm(stream, hin, W, nullptr, xw, NN, 256, Kin, 0, EPI_NONE);    // xw = h @ W
            gat_attn_coef_kernel<<<(NN * 4 + 255) / 256, 256, 0, stream>>>(xw, as_, ad_, al, ar, NN);
            fillu(stream, nmax, ORD_NEG_INF, (size_t)NN * 4);
            fillf(stream, nsum, 0.f, (size_t)NN * 4);
            fillf(stream, outbuf, 0.f, (size_t)NN * 256);
            gat_edge_max_kernel<<<(Et4 + 255) / 256, 256, 0, stream>>>(src, dst, E, NN, al, ar, nmax);
            gat_edge_sum_kernel<<<(Et4 + 255) / 256, 256, 0, stream>>>(src, dst, E, NN, al, ar, nmax, nsum);
            gat_edge_agg_kernel<<<(Et4 + 255) / 256, 256, 0, stream>>>(src, dst, E, NN, al, ar, nmax, nsum, xw, outbuf);
            gat_finalize_kernel<<<(unsigned)(((size_t)NN * 256 + 255) / 256), 256, 0, stream>>>(outbuf, bias, (size_t)NN * 256);
        }
    }

    // ---------------- semantic attention -> attn_applied (d_out tail) ------
    semantic_kernel<<<(NN + 255) / 256, 256, 0, stream>>>(embs, 0, 3, sem_w,       sem_b,     attn_out, 0,   NN);
    semantic_kernel<<<(NN + 255) / 256, 256, 0, stream>>>(embs, 3, 2, sem_w + 256, sem_b + 1, attn_out, 256, NN);

    // ---------------- einsum btc,cd->btd  (512 x 2500 x 512) ---------------
    gemm(stream, x, attn_out, nullptr, xt, 512, 512, 2500, 0, EPI_NONE);
    transpose_bt_kernel<<<(512 * 512 + 255) / 256, 256, 0, stream>>>(xt, day);

    // ---------------- GRU gate-input precompute ----------------------------
    gemm(stream, day, Wih,             bih,       gif, 512, 900, 512, 1, EPI_BIAS);
    gemm(stream, day, Wih + 900 * 512, bih + 900, gib, 512, 900, 512, 1, EPI_BIAS);

    // Forward GRU: single scan (causal -> shared across all prefixes)
    fillf(stream, hstate, 0.f, 32UL * 300);
    for (int t = 0; t < 16; ++t) {
        gemm(stream, hstate, Whh, bhh, ghf, 32, 900, 300, 1, EPI_BIAS);
        gru_fwd_step_kernel<<<(32 * 300 + 255) / 256, 256, 0, stream>>>(gif, ghf, hstate, fh, t);
    }

    // Backward GRUs: 16 chains batched into one GEMM per step
    fillf(stream, Hb, 0.f, 512UL * 300);
    for (int s = 0; s < 16; ++s) {
        gemm(stream, Hb, Whh + 900 * 300, bhh + 900, ghb, 512, 900, 300, 1, EPI_BIAS);
        gru_bwd_step_kernel<<<(16 * 32 * 300 + 255) / 256, 256, 0, stream>>>(gib, ghb, Hb, rh, s);
    }

    // ---------------- temporal attention + projections ---------------------
    time_attn_kernel<<<512, 128, 0, stream>>>(fh, rh, attn_w, attn_b, ht);
    gemm(stream, ht,    aout_w, aout_b, houts, 512, 512, 1200, 1, EPI_BIAS);
    gemm(stream, houts, out_w,  out_b,  ytmp,  512, 500, 512,  1, EPI_BIAS_SIGMOID);
    out_permute_kernel<<<(512 * 500 + 255) / 256, 256, 0, stream>>>(ytmp, y);
}
